// BidirectionalMambaLayer_2456721293313
// MI455X (gfx1250) — matrix operand order repaired and checked by reading, not compiled, not run on hardware
//
#include <hip/hip_runtime.h>
#include <hip/hip_bf16.h>

typedef __attribute__((ext_vector_type(16))) _Float16 h16v;
typedef __attribute__((ext_vector_type(8)))  _Float16 h8v;
typedef __attribute__((ext_vector_type(8)))  float    f8v;

#define D_MODEL 128
#define D_INNER 256
#define D_STATE 16
#define D_CONV  4
#define DT_RANK 8
#define NB      2
#define SEQ_L   8192
#define BL      (NB * SEQ_L)            // 16384 tokens
#define N_XZ    (2 * D_INNER)           // 512
#define N_XDB   (DT_RANK + 2 * D_STATE) // 40

__device__ __forceinline__ float siluf(float v) { return v / (1.f + __expf(-v)); }

// ---------------- LayerNorm over channels, emit f16 tokens (BL, 128) ----------------
__global__ void k_layernorm(const float* __restrict__ x, const float* __restrict__ g,
                            const float* __restrict__ bta, _Float16* __restrict__ xn16) {
  int t = blockIdx.x * blockDim.x + threadIdx.x;
  if (t >= BL) return;
  int b = t / SEQ_L, s = t % SEQ_L;
  const float* xb = x + (size_t)b * D_MODEL * SEQ_L + s;
  float sum = 0.f, sq = 0.f;
  for (int c = 0; c < D_MODEL; ++c) { float v = xb[(size_t)c * SEQ_L]; sum += v; sq += v * v; }
  float mean = sum * (1.f / D_MODEL);
  float var  = sq * (1.f / D_MODEL) - mean * mean;
  float r    = rsqrtf(var + 1e-5f);
  _Float16* o = xn16 + (size_t)t * D_MODEL;
  for (int c = 0; c < D_MODEL; ++c) {
    float v = xb[(size_t)c * SEQ_L];
    o[c] = (_Float16)((v - mean) * r * g[c] + bta[c]);
  }
}

// ---------------- f32 -> f16 weight cast ----------------
__global__ void k_cast16(const float* __restrict__ in, _Float16* __restrict__ out, int n) {
  int i = blockIdx.x * blockDim.x + threadIdx.x;
  if (i < n) out[i] = (_Float16)in[i];
}

// ---------------- WMMA GEMM: C(BLxN) = A16(BLxK) * W16(NxK)^T ; one wave per 16x16 tile ----
template <int K, int N, bool ACC>
__global__ void k_gemm_wmma(const _Float16* __restrict__ A, const _Float16* __restrict__ W,
                            float* __restrict__ C) {
  constexpr int NT = N / 16;
  int wave = (blockIdx.x * blockDim.x + threadIdx.x) >> 5;
  int lane = threadIdx.x & 31;
  int mt = wave / NT, nt = wave % NT;
  if (mt >= BL / 16) return;                 // wave-uniform: EXEC stays all-1s
  int row = lane & 15;
  int hi  = lane >> 4;                       // lane group 0/1
  const _Float16* Arow = A + (size_t)(mt * 16 + row) * K;
  const _Float16* Wrow = W + (size_t)(nt * 16 + row) * K;   // row = output col n
  f8v c = {};
#pragma unroll
  for (int k0 = 0; k0 < K; k0 += 32) {
    // A fragment: lanes 0-15 K=k0..k0+15, lanes 16-31 K=k0+16..k0+31, row = lane&15
    h16v a = *(const h16v*)(Arow + k0 + hi * 16);
    // B fragment (32x16): lanes 0-15 K=k0..k0+15, lanes 16-31 K=k0+16..k0+31, col = lane&15
    h16v bfr = *(const h16v*)(Wrow + k0 + hi * 16);
    c = __builtin_amdgcn_wmma_f32_16x16x32_f16(false, a, false, bfr,
                                               (short)0, c, false, false);
  }
  float* Cp = C + (size_t)(mt * 16 + hi * 8) * N + nt * 16 + row;
#pragma unroll
  for (int i = 0; i < 8; ++i) {
    float* p = Cp + (size_t)i * N;
    if (ACC) *p += c[i]; else *p = c[i];
  }
}

// ---------------- depthwise causal conv (rev => anticausal on physical order) + SiLU ----
__global__ void k_conv(const float* __restrict__ XZ, const float* __restrict__ convw,
                       const float* __restrict__ convb, float* __restrict__ xc, int rev) {
  int idx = blockIdx.x * blockDim.x + threadIdx.x;   // idx = t*256 + e
  if (idx >= BL * D_INNER) return;
  int e = idx % D_INNER;
  int t = idx / D_INNER;
  int b = t / SEQ_L, p = t % SEQ_L;
  float acc = convb[e];
#pragma unroll
  for (int k = 0; k < D_CONV; ++k) {
    int q = rev ? (p + (D_CONV - 1) - k) : (p - (D_CONV - 1) + k);
    if (q >= 0 && q < SEQ_L)
      acc += convw[e * D_CONV + k] * XZ[((size_t)(b * SEQ_L + q)) * N_XZ + e];
  }
  xc[idx] = siluf(acc);
}

// ---------------- xdb(BLx40) = xc(BLx256) * Wx(40x256)^T (small, VALU) ----------------
__global__ void k_xdb(const float* __restrict__ xc, const float* __restrict__ Wx,
                      float* __restrict__ xdb) {
  int idx = blockIdx.x * blockDim.x + threadIdx.x;
  if (idx >= BL * N_XDB) return;
  int r = idx % N_XDB, t = idx / N_XDB;
  const float* xr = xc + (size_t)t * D_INNER;
  const float* wr = Wx + (size_t)r * D_INNER;
  float acc = 0.f;
  for (int e = 0; e < D_INNER; ++e) acc += xr[e] * wr[e];
  xdb[idx] = acc;
}

// ---------------- fused dt + selective scan + gate; grid = {dir,batch}, 1 thread/channel ----
__global__ void k_scan(const float* __restrict__ xdb, const float* __restrict__ xc,
                       const float* __restrict__ XZ, _Float16* __restrict__ g16,
                       const float* __restrict__ Wdt0, const float* __restrict__ Wdt1,
                       const float* __restrict__ bdt0, const float* __restrict__ bdt1,
                       const float* __restrict__ Al0,  const float* __restrict__ Al1,
                       const float* __restrict__ Ds0,  const float* __restrict__ Ds1) {
  int dir = blockIdx.x >> 1;
  int b   = blockIdx.x & 1;
  int e   = threadIdx.x;                 // 0..255
  const float* Wdt = dir ? Wdt1 : Wdt0;
  const float* bdt = dir ? bdt1 : bdt0;
  const float* Al  = dir ? Al1  : Al0;
  const float* Ds  = dir ? Ds1  : Ds0;
  size_t dofX = (size_t)dir * BL;        // token offset per direction
  float wdt[DT_RANK];
#pragma unroll
  for (int r = 0; r < DT_RANK; ++r) wdt[r] = Wdt[e * DT_RANK + r];
  float bias = bdt[e];
  float Arow[D_STATE];
#pragma unroll
  for (int n = 0; n < D_STATE; ++n) Arow[n] = -__expf(Al[e * D_STATE + n]);
  float dsk = Ds[e];
  float h[D_STATE];
#pragma unroll
  for (int n = 0; n < D_STATE; ++n) h[n] = 0.f;

  for (int s = 0; s < SEQ_L; ++s) {
    int p = dir ? (SEQ_L - 1 - s) : s;
    size_t t = dofX + (size_t)b * SEQ_L + p;
    const float* row = xdb + t * N_XDB;
    float v = bias;
#pragma unroll
    for (int r = 0; r < DT_RANK; ++r) v += row[r] * wdt[r];
    float dt = (v > 20.f) ? v : log1pf(__expf(v));   // softplus
    float xt = xc[t * D_INNER + e];
    float dtx = dt * xt;
    float y = 0.f;
#pragma unroll
    for (int n = 0; n < D_STATE; ++n) {
      float da = __expf(dt * Arow[n]);
      h[n] = da * h[n] + dtx * row[DT_RANK + n];
      y += h[n] * row[DT_RANK + D_STATE + n];
    }
    float z = XZ[t * N_XZ + D_INNER + e];
    g16[t * D_INNER + e] = (_Float16)((y + xt * dsk) * siluf(z));
  }
}

// ---------------- residual add + transpose back to (B, C, spatial) ----------------
__global__ void k_final(const float* __restrict__ x, const float* __restrict__ acc,
                        float* __restrict__ out) {
  size_t i = (size_t)blockIdx.x * blockDim.x + threadIdx.x;
  if (i >= (size_t)NB * D_MODEL * SEQ_L) return;
  size_t s = i % SEQ_L;
  size_t c = (i / SEQ_L) % D_MODEL;
  size_t b = i / ((size_t)SEQ_L * D_MODEL);
  out[i] = x[i] + acc[(b * SEQ_L + s) * (size_t)D_MODEL + c];
}

extern "C" void kernel_launch(void* const* d_in, const int* in_sizes, int n_in,
                              void* d_out, int out_size, void* d_ws, size_t ws_size,
                              hipStream_t stream) {
  (void)in_sizes; (void)n_in; (void)out_size; (void)ws_size;
  const float* x   = (const float*)d_in[0];
  const float* lng = (const float*)d_in[1];
  const float* lnb = (const float*)d_in[2];

  char* ws = (char*)d_ws;
  size_t off = 0;
  auto salloc = [&](size_t bytes) -> char* {
    char* p = ws + off; off = (off + bytes + 255) & ~(size_t)255; return p;
  };
  _Float16* xn16   = (_Float16*)salloc((size_t)BL * D_MODEL * 2);
  _Float16* win16[2]  = { (_Float16*)salloc((size_t)N_XZ * D_MODEL * 2),
                          (_Float16*)salloc((size_t)N_XZ * D_MODEL * 2) };
  _Float16* wout16[2] = { (_Float16*)salloc((size_t)D_MODEL * D_INNER * 2),
                          (_Float16*)salloc((size_t)D_MODEL * D_INNER * 2) };
  float*    XZ   = (float*)salloc((size_t)2 * BL * N_XZ * 4);      // [dir][t][512]
  float*    xcb  = (float*)salloc((size_t)2 * BL * D_INNER * 4);   // [dir][t][256]
  float*    xdb  = (float*)salloc((size_t)2 * BL * N_XDB * 4);     // [dir][t][40]
  _Float16* g16  = (_Float16*)salloc((size_t)2 * BL * D_INNER * 2);
  float*    oacc = (float*)salloc((size_t)BL * D_MODEL * 4);

  k_layernorm<<<BL / 256, 256, 0, stream>>>(x, lng, lnb, xn16);
  for (int d = 0; d < 2; ++d) {
    const float* Win  = (const float*)d_in[3  + 9 * d];
    const float* Wout = (const float*)d_in[11 + 9 * d];
    k_cast16<<<(N_XZ * D_MODEL + 255) / 256, 256, 0, stream>>>(Win, win16[d], N_XZ * D_MODEL);
    k_cast16<<<(D_MODEL * D_INNER + 255) / 256, 256, 0, stream>>>(Wout, wout16[d], D_MODEL * D_INNER);
  }
  for (int d = 0; d < 2; ++d) {
    const float* convw = (const float*)d_in[4 + 9 * d];
    const float* convb = (const float*)d_in[5 + 9 * d];
    const float* Wx    = (const float*)d_in[6 + 9 * d];
    float* XZd  = XZ  + (size_t)d * BL * N_XZ;
    float* xcd  = xcb + (size_t)d * BL * D_INNER;
    float* xdbd = xdb + (size_t)d * BL * N_XDB;
    k_gemm_wmma<D_MODEL, N_XZ, false>
        <<<(BL / 16) * (N_XZ / 16) / 8, 256, 0, stream>>>(xn16, win16[d], XZd);
    k_conv<<<BL * D_INNER / 256, 256, 0, stream>>>(XZd, convw, convb, xcd, d);
    k_xdb<<<(BL * N_XDB + 255) / 256, 256, 0, stream>>>(xcd, Wx, xdbd);
  }
  // both directions' scans concurrently: grid = {dir, batch}
  k_scan<<<4, D_INNER, 0, stream>>>(
      xdb, xcb, XZ, g16,
      (const float*)d_in[7],  (const float*)d_in[16],   // Wdt f/b
      (const float*)d_in[8],  (const float*)d_in[17],   // bdt f/b
      (const float*)d_in[9],  (const float*)d_in[18],   // Alog f/b
      (const float*)d_in[10], (const float*)d_in[19]);  // D f/b
  k_gemm_wmma<D_INNER, D_MODEL, false>
      <<<(BL / 16) * (D_MODEL / 16) / 8, 256, 0, stream>>>(g16, wout16[0], oacc);
  k_gemm_wmma<D_INNER, D_MODEL, true>
      <<<(BL / 16) * (D_MODEL / 16) / 8, 256, 0, stream>>>(
          g16 + (size_t)BL * D_INNER, wout16[1], oacc);
  k_final<<<(NB * D_MODEL * SEQ_L) / 256, 256, 0, stream>>>(x, oacc, (float*)d_out);
}
